// NeuroMambaBCE_4793183502533
// MI455X (gfx1250) — matrix operand order repaired and checked by reading, compiled, not run on hardware
//
#include <hip/hip_runtime.h>
#include <hip/hip_bf16.h>
#include <math.h>

// ---------------- dims ----------------
#define BSZ      2
#define SEQ      512
#define RTOT     (BSZ * SEQ)        // 1024 token rows
#define DM       272                // D_MODEL
#define DMP      288                // D_MODEL K-padded to mult of 32
#define DI       544                // D_INNER
#define DI2      1088               // 2*D_INNER
#define DSTATE   32
#define DTRANK   17
#define NDBC     96                 // 81 -> padded to 96 (6 n-tiles)

typedef _Float16 half_t;
typedef __attribute__((ext_vector_type(16))) _Float16 v16h;
typedef __attribute__((ext_vector_type(8)))  float    v8f;
typedef __attribute__((ext_vector_type(4)))  unsigned u32x4;

union FragH { v16h v; u32x4 u[2]; };

// ---------------- helpers ----------------
__device__ __forceinline__ float siluf(float x) {
    return x / (1.0f + __expf(-x));
}
__device__ __forceinline__ float softplusf(float x) {
    return (x > 20.0f) ? x : log1pf(__expf(x));
}
// block reduction for blockDim.x <= 512 (wave32), smem needs >= 16 floats
__device__ __forceinline__ float blockReduceSum(float v, float* smem) {
    __syncthreads();
    #pragma unroll
    for (int o = 16; o > 0; o >>= 1) v += __shfl_xor(v, o, 32);
    int wid = threadIdx.x >> 5, lid = threadIdx.x & 31;
    int nw = (blockDim.x + 31) >> 5;
    if (lid == 0) smem[wid] = v;
    __syncthreads();
    if (threadIdx.x < 32) {
        float s = (threadIdx.x < nw) ? smem[threadIdx.x] : 0.0f;
        #pragma unroll
        for (int o = 16; o > 0; o >>= 1) s += __shfl_xor(s, o, 32);
        if (threadIdx.x == 0) smem[0] = s;
    }
    __syncthreads();
    return smem[0];
}

// ---------------- kernels ----------------

// RMSNorm a row of x_cur -> f16 K-padded, written in both normal and flipped row order.
__global__ __launch_bounds__(DMP)
void rmsnorm_f16_kernel(const float* __restrict__ x, const float* __restrict__ w,
                        half_t* __restrict__ out, half_t* __restrict__ outf) {
    __shared__ float sm[16];
    int r = blockIdx.x;              // b*SEQ + l
    int t = threadIdx.x;             // 0..287
    float v = (t < DM) ? x[(size_t)r * DM + t] : 0.0f;
    float ss = blockReduceSum(v * v, sm);
    float scale = rsqrtf(ss / (float)DM + 1e-5f);
    int b = r / SEQ, l = r % SEQ;
    int rf = b * SEQ + (SEQ - 1 - l);
    half_t h = (half_t)((t < DM) ? v * scale * w[t] : 0.0f);
    out [(size_t)r  * DMP + t] = h;
    outf[(size_t)rf * DMP + t] = h;
}

// convert f32 weight (N x K) to zero-padded f16 (Np x Kp)
__global__ __launch_bounds__(256)
void cvt_pad_f16_kernel(const float* __restrict__ src, half_t* __restrict__ dst,
                        int N, int K, int Np, int Kp) {
    int idx = blockIdx.x * blockDim.x + threadIdx.x;
    if (idx >= Np * Kp) return;
    int n = idx / Kp, k = idx - n * Kp;
    float v = (n < N && k < K) ? src[(size_t)n * K + k] : 0.0f;
    dst[idx] = (half_t)v;
}

// C (R x N) f32 = A (R x Kp) f16  @  W^T, with W (N x Kp) f16 row-major.
// One wave32 per 16x16 output tile; K in chunks of 32 through v_wmma_f32_16x16x32_f16.
__global__ __launch_bounds__(32)
void wmma_gemm_kernel(const half_t* __restrict__ A, const half_t* __restrict__ W,
                      float* __restrict__ C, int Kp, int N) {
    int lane = threadIdx.x;
    int mt = blockIdx.x, nt = blockIdx.y;
    int lo16 = lane & 15, hi = lane >> 4;
    const half_t* pa = A + (size_t)(mt * 16 + lo16) * Kp;
    const half_t* pb = W + (size_t)(nt * 16 + lo16) * Kp;
    v8f acc = {};
    for (int kb = 0; kb < Kp; kb += 32) {
        FragH a, b;
        // A frag (16-bit 16x32 layout): two 8-half runs at kb+hi*8 and kb+16+hi*8
        a.u[0] = *reinterpret_cast<const u32x4*>(pa + kb + hi * 8);
        a.u[1] = *reinterpret_cast<const u32x4*>(pa + kb + 16 + hi * 8);
        // B frag (32x16): two 8-half runs of W row n at kb+hi*8 and kb+16+hi*8
        b.u[0] = *reinterpret_cast<const u32x4*>(pb + kb + hi * 8);
        b.u[1] = *reinterpret_cast<const u32x4*>(pb + kb + 16 + hi * 8);
        acc = __builtin_amdgcn_wmma_f32_16x16x32_f16(
            false, a.v, false, b.v, (short)0, acc, false, false);
    }
    // D layout: lane&15 = N col, VGPR v -> M row = v + 8*hi
    float* pc = C + (size_t)(mt * 16 + hi * 8) * N + (nt * 16 + lo16);
    #pragma unroll
    for (int v = 0; v < 8; ++v) pc[(size_t)v * N] = acc[v];
}

// causal depthwise conv (k=4) + SiLU; input xin = first DI cols of xz
__global__ __launch_bounds__(DI)
void conv_silu_kernel(const float* __restrict__ xz, const float* __restrict__ cw,
                      const float* __restrict__ cb, float* __restrict__ xc,
                      half_t* __restrict__ xc16) {
    int r = blockIdx.x;
    int d = threadIdx.x;             // 0..543
    int b = r / SEQ, l = r % SEQ;
    float acc = cb[d];
    #pragma unroll
    for (int k = 0; k < 4; ++k) {
        int ls = l - 3 + k;
        if (ls >= 0) acc += cw[d * 4 + k] * xz[(size_t)(b * SEQ + ls) * DI2 + d];
    }
    float s = siluf(acc);
    xc  [(size_t)r * DI + d] = s;
    xc16[(size_t)r * DI + d] = (half_t)s;
}

// dt = softplus(dbc[:, :17] @ dtw^T + dtb)    (K=17: scalar path)
__global__ __launch_bounds__(DI)
void dt_kernel(const float* __restrict__ dbc, const float* __restrict__ dtw,
               const float* __restrict__ dtb, float* __restrict__ dt) {
    __shared__ float row[DTRANK];
    int r = blockIdx.x, d = threadIdx.x;
    if (d < DTRANK) row[d] = dbc[(size_t)r * NDBC + d];
    __syncthreads();
    float acc = dtb[d];
    #pragma unroll
    for (int j = 0; j < DTRANK; ++j) acc += row[j] * dtw[d * DTRANK + j];
    dt[(size_t)r * DI + d] = softplusf(acc);
}

// selective scan: lane = state (wave32 == D_STATE), 8 channels per wave,
// h-update in registers, h.C reduced with shfl_xor tree.
__global__ __launch_bounds__(32)
void scan_kernel(const float* __restrict__ dt, const float* __restrict__ xc,
                 const float* __restrict__ dbc, const float* __restrict__ a_log,
                 float* __restrict__ ys) {
    int s  = threadIdx.x;            // state 0..31
    int c0 = blockIdx.x * 8;         // channel group
    int b  = blockIdx.y;
    float A[8], h[8];
    #pragma unroll
    for (int c = 0; c < 8; ++c) {
        A[c] = -__expf(a_log[(size_t)(c0 + c) * DSTATE + s]);
        h[c] = 0.0f;
    }
    for (int l = 0; l < SEQ; ++l) {
        size_t r = (size_t)b * SEQ + l;
        float Bv = dbc[r * NDBC + DTRANK + s];
        float Cv = dbc[r * NDBC + DTRANK + DSTATE + s];
        if (l + 1 < SEQ) {
            __builtin_prefetch(dt + (r + 1) * DI + c0, 0, 1);
            __builtin_prefetch(dbc + (r + 1) * NDBC, 0, 1);
        }
        #pragma unroll
        for (int c = 0; c < 8; ++c) {
            float dtv = dt[r * DI + c0 + c];
            float xv  = xc[r * DI + c0 + c];
            h[c] = __expf(dtv * A[c]) * h[c] + (dtv * xv) * Bv;
            float p = h[c] * Cv;
            #pragma unroll
            for (int o = 16; o > 0; o >>= 1) p += __shfl_xor(p, o, 32);
            if (s == 0) ys[r * DI + c0 + c] = p;
        }
    }
}

// yo = (ys + Dskip*xc) * silu(z), cast to f16 for the out-projection
__global__ __launch_bounds__(DI)
void gate_kernel(const float* __restrict__ ys, const float* __restrict__ xc,
                 const float* __restrict__ xz, const float* __restrict__ dskip,
                 half_t* __restrict__ yo16) {
    int r = blockIdx.x, d = threadIdx.x;
    float z = xz[(size_t)r * DI2 + DI + d];
    float v = (ys[(size_t)r * DI + d] + dskip[d] * xc[(size_t)r * DI + d]) * siluf(z);
    yo16[(size_t)r * DI + d] = (half_t)v;
}

// attn = rmsnorm(y1 - lam_full*flip(y2), n2w); x_cur += attn*(1-lam_init)
__global__ __launch_bounds__(DMP)
void combine_kernel(float* __restrict__ x_cur, const float* __restrict__ y1,
                    const float* __restrict__ y2f, const float* __restrict__ lam_q,
                    const float* __restrict__ n2w, float lam_init) {
    __shared__ float sm[16];
    int r = blockIdx.x, t = threadIdx.x;
    float lv = (t < DM) ? lam_q[t] : 0.0f;
    float lam_sum = blockReduceSum(lv, sm);
    float lam_full = 1.0f / (1.0f + __expf(-lam_sum)) + lam_init;
    int b = r / SEQ, l = r % SEQ;
    size_t rf = (size_t)b * SEQ + (SEQ - 1 - l);
    float d = 0.0f;
    if (t < DM) d = y1[(size_t)r * DM + t] - lam_full * y2f[rf * DM + t];
    float ss = blockReduceSum(d * d, sm);
    float scale = rsqrtf(ss / (float)DM + 1e-5f);
    if (t < DM) x_cur[(size_t)r * DM + t] += (d * scale * n2w[t]) * (1.0f - lam_init);
}

__global__ __launch_bounds__(DMP)
void mean_kernel(const float* __restrict__ x_cur, float* __restrict__ xm) {
    int b = blockIdx.x, d = threadIdx.x;
    if (d >= DM) return;
    float s = 0.0f;
    for (int l = 0; l < SEQ; ++l) s += x_cur[(size_t)(b * SEQ + l) * DM + d];
    xm[b * DM + d] = s / (float)SEQ;
}

// x2 = relu(xm @ mlp_w^T + mlp_b); logits = [x2, y] @ predict_w^T + pb
// d_out layout: [logits(B)] then [x2(B*DM)]
__global__ __launch_bounds__(DMP)
void head_kernel(const float* __restrict__ xm, const float* __restrict__ mlp_w,
                 const float* __restrict__ mlp_b, const float* __restrict__ pw,
                 const float* __restrict__ pb, const float* __restrict__ y,
                 float* __restrict__ out) {
    __shared__ float sm[16];
    __shared__ float xs[DM];
    int b = blockIdx.x, t = threadIdx.x;
    if (t < DM) xs[t] = xm[b * DM + t];
    __syncthreads();
    float x2 = 0.0f;
    if (t < DM) {
        float acc = mlp_b[t];
        for (int j = 0; j < DM; ++j) acc += mlp_w[(size_t)t * DM + j] * xs[j];
        x2 = fmaxf(acc, 0.0f);
        out[BSZ + b * DM + t] = x2;
    }
    float part = (t < DM) ? pw[t] * x2 : 0.0f;
    float s = blockReduceSum(part, sm);
    if (t == 0) out[b] = s + pw[DM] * y[b] + pb[0];
}

// ---------------- host ----------------
extern "C" void kernel_launch(void* const* d_in, const int* in_sizes, int n_in,
                              void* d_out, int out_size, void* d_ws, size_t ws_size,
                              hipStream_t stream) {
    (void)in_sizes; (void)n_in; (void)out_size; (void)ws_size;
    const float* x          = (const float*)d_in[0];
    const float* y          = (const float*)d_in[1];
    const float* in_proj_w  = (const float*)d_in[2];
    const float* conv_w     = (const float*)d_in[3];
    const float* conv_b     = (const float*)d_in[4];
    const float* x_proj_w   = (const float*)d_in[5];
    const float* dt_proj_w  = (const float*)d_in[6];
    const float* dt_proj_b  = (const float*)d_in[7];
    const float* A_log      = (const float*)d_in[8];
    const float* Dskip      = (const float*)d_in[9];
    const float* out_proj_w = (const float*)d_in[10];
    const float* lambda_q1  = (const float*)d_in[11];
    const float* norm1_w    = (const float*)d_in[12];
    const float* norm2_w    = (const float*)d_in[13];
    const float* mlp_w      = (const float*)d_in[14];
    const float* mlp_b      = (const float*)d_in[15];
    const float* predict_w  = (const float*)d_in[16];
    const float* predict_b  = (const float*)d_in[17];

    char* ws = (char*)d_ws;
    size_t off = 0;
    auto carve = [&](size_t bytes) -> void* {
        void* p = ws + off;
        off = (off + bytes + 255) & ~(size_t)255;
        return p;
    };
    float*  x_cur  = (float*) carve((size_t)RTOT * DM  * 4);
    half_t* xn16   = (half_t*)carve((size_t)RTOT * DMP * 2);
    half_t* xn16f  = (half_t*)carve((size_t)RTOT * DMP * 2);
    half_t* w16_in = (half_t*)carve((size_t)DI2  * DMP * 2);
    half_t* w16_x  = (half_t*)carve((size_t)NDBC * DI  * 2);
    half_t* w16_o  = (half_t*)carve((size_t)DM   * DI  * 2);
    float*  xz     = (float*) carve((size_t)RTOT * DI2 * 4);
    float*  xc     = (float*) carve((size_t)RTOT * DI  * 4);
    half_t* xc16   = (half_t*)carve((size_t)RTOT * DI  * 2);
    float*  dbc    = (float*) carve((size_t)RTOT * NDBC* 4);
    float*  dtb_   = (float*) carve((size_t)RTOT * DI  * 4);
    float*  ysb    = (float*) carve((size_t)RTOT * DI  * 4);
    half_t* yo16   = (half_t*)carve((size_t)RTOT * DI  * 2);
    float*  y1     = (float*) carve((size_t)RTOT * DM  * 4);
    float*  y2f    = (float*) carve((size_t)RTOT * DM  * 4);
    float*  xm     = (float*) carve((size_t)BSZ  * DM  * 4);

    hipMemcpyAsync(x_cur, x, (size_t)RTOT * DM * sizeof(float),
                   hipMemcpyDeviceToDevice, stream);

    for (int i = 0; i < 12; ++i) {
        float lam_init = 0.8f - 0.6f * expf(-0.3f * (float)i);
        rmsnorm_f16_kernel<<<RTOT, DMP, 0, stream>>>(x_cur, norm1_w + i * DM, xn16, xn16f);

        for (int dir = 0; dir < 2; ++dir) {
            int li = i * 2 + dir;
            const half_t* xnA = dir ? xn16f : xn16;
            float* yout = dir ? y2f : y1;

            // in_proj: W (1088 x 272) -> f16 padded K=288
            cvt_pad_f16_kernel<<<(DI2 * DMP + 255) / 256, 256, 0, stream>>>(
                in_proj_w + (size_t)li * DI2 * DM, w16_in, DI2, DM, DI2, DMP);
            wmma_gemm_kernel<<<dim3(RTOT / 16, DI2 / 16), 32, 0, stream>>>(
                xnA, w16_in, xz, DMP, DI2);

            conv_silu_kernel<<<RTOT, DI, 0, stream>>>(
                xz, conv_w + (size_t)li * DI * 4, conv_b + (size_t)li * DI, xc, xc16);

            // x_proj: W (81 x 544) -> f16 padded N=96
            cvt_pad_f16_kernel<<<(NDBC * DI + 255) / 256, 256, 0, stream>>>(
                x_proj_w + (size_t)li * 81 * DI, w16_x, 81, DI, NDBC, DI);
            wmma_gemm_kernel<<<dim3(RTOT / 16, NDBC / 16), 32, 0, stream>>>(
                xc16, w16_x, dbc, DI, NDBC);

            dt_kernel<<<RTOT, DI, 0, stream>>>(
                dbc, dt_proj_w + (size_t)li * DI * DTRANK, dt_proj_b + (size_t)li * DI, dtb_);

            scan_kernel<<<dim3(DI / 8, BSZ), 32, 0, stream>>>(
                dtb_, xc, dbc, A_log + (size_t)li * DI * DSTATE, ysb);

            gate_kernel<<<RTOT, DI, 0, stream>>>(
                ysb, xc, xz, Dskip + (size_t)li * DI, yo16);

            // out_proj: W (272 x 544)
            cvt_pad_f16_kernel<<<(DM * DI + 255) / 256, 256, 0, stream>>>(
                out_proj_w + (size_t)li * DM * DI, w16_o, DM, DI, DM, DI);
            wmma_gemm_kernel<<<dim3(RTOT / 16, DM / 16), 32, 0, stream>>>(
                yo16, w16_o, yout, DI, DM);
        }

        combine_kernel<<<RTOT, DMP, 0, stream>>>(
            x_cur, y1, y2f, lambda_q1 + i * DM, norm2_w + i * DM, lam_init);
    }

    mean_kernel<<<BSZ, DMP, 0, stream>>>(x_cur, xm);
    head_kernel<<<BSZ, DMP, 0, stream>>>(xm, mlp_w, mlp_b, predict_w, predict_b, y,
                                         (float*)d_out);
}
